// NonAutoregressiveEncoder_44349832298867
// MI455X (gfx1250) — matrix operand order repaired and checked by reading, not compiled, not run on hardware
//
#include <hip/hip_runtime.h>
#include <hip/hip_bf16.h>

typedef __attribute__((ext_vector_type(16))) _Float16 v16h;
typedef __attribute__((ext_vector_type(8)))  _Float16 v8h;
typedef __attribute__((ext_vector_type(8)))  float    v8f;

#define D 64
static constexpr float BN_EPS = 1e-5f;

__device__ __forceinline__ float sigmoidf_(float z) { return 1.0f / (1.0f + __expf(-z)); }
__device__ __forceinline__ float siluf_(float z)    { return z * sigmoidf_(z); }

// ---------------- WMMA fragment loaders (CDNA5 wave32, 16x16x32 f16) ----------------
// A (16x32 f16), ISA 7.12.2: lanes 0-15 hold row=lane with K = k0+{0..7} in elems 0-7 and
// K=k0+{16..23} in elems 8-15; lanes 16-31 hold the same row with K shifted by 8.
__device__ __forceinline__ v16h load_a_frag(const float* __restrict__ X, int row, int k0, int lane) {
  const float* p = X + (size_t)row * D + k0 + ((lane >> 4) << 3);
  const float4 q0 = *(const float4*)(p);
  const float4 q1 = *(const float4*)(p + 4);
  const float4 q2 = *(const float4*)(p + 16);
  const float4 q3 = *(const float4*)(p + 20);
  v16h a;
  a[0]  = (_Float16)q0.x; a[1]  = (_Float16)q0.y; a[2]  = (_Float16)q0.z; a[3]  = (_Float16)q0.w;
  a[4]  = (_Float16)q1.x; a[5]  = (_Float16)q1.y; a[6]  = (_Float16)q1.z; a[7]  = (_Float16)q1.w;
  a[8]  = (_Float16)q2.x; a[9]  = (_Float16)q2.y; a[10] = (_Float16)q2.z; a[11] = (_Float16)q2.w;
  a[12] = (_Float16)q3.x; a[13] = (_Float16)q3.y; a[14] = (_Float16)q3.z; a[15] = (_Float16)q3.w;
  return a;
}

// B (32x16 f16) from pre-transposed f16 weights Wt[out][in] (64x64, row-major):
// lanes 0-15 hold col = n0+lane with K = k0+{0..7,16..23}; lanes 16-31 hold K = k0+{8..15,24..31}.
// Per-lane data is two runs of 8 contiguous halves, 16 apart -> two 16-byte loads, no converts.
__device__ __forceinline__ v16h load_b_frag(const _Float16* __restrict__ Wt, int n0, int k0, int lane) {
  const _Float16* p = Wt + (size_t)(n0 + (lane & 15)) * D + k0 + ((lane >> 4) << 3);
  const v8h lo = *(const v8h*)(p);
  const v8h hi = *(const v8h*)(p + 16);
  return __builtin_shufflevector(lo, hi, 0, 1, 2, 3, 4, 5, 6, 7, 8, 9, 10, 11, 12, 13, 14, 15);
}

// ---------------- zero scratch ----------------
__global__ void zero_kernel(float* __restrict__ p, size_t n) {
  size_t i = (size_t)blockIdx.x * blockDim.x + threadIdx.x;
  const size_t stride = (size_t)gridDim.x * blockDim.x;
  for (; i < n; i += stride) p[i] = 0.0f;
}

// ---------------- per-layer weight prep: f16 + transpose to [out][in] ----------------
// mats 0..3 = v_w[l][k], mat 4 = e_w[l]
__global__ void __launch_bounds__(256)
weight_prep_kernel(const float* __restrict__ vw, const float* __restrict__ ew,
                   _Float16* __restrict__ wt) {
  const int tid = blockIdx.x * 256 + threadIdx.x;
  if (tid >= 5 * D * D) return;
  const int m   = tid >> 12;        // matrix index
  const int idx = tid & (D * D - 1);
  const int col = idx >> 6;         // output dim e
  const int row = idx & (D - 1);    // input  dim d
  const float* W = (m < 4) ? (vw + (size_t)m * D * D) : ew;
  wt[(size_t)m * D * D + (size_t)col * D + row] = (_Float16)W[(size_t)row * D + col];
}

// ---------------- node projections: xp[k] = x @ v_w[k] + v_b[k], k=0..3 ----------------
__global__ void __launch_bounds__(128)
node_proj_kernel(const float* __restrict__ x,
                 const _Float16* __restrict__ wt,  // [4][D][D] transposed f16
                 const float* __restrict__ vb,     // [4][D]
                 float* __restrict__ xp,           // [4][N][D]
                 int N) {
  const int k    = threadIdx.x >> 5;   // projection index (4 waves/block)
  const int lane = threadIdx.x & 31;
  const int row0 = blockIdx.x * 16;
  const bool full = (row0 + 16 <= N);
  const int arow = full ? (row0 + (lane & 15)) : min(row0 + (lane & 15), N - 1);
  const _Float16* W = wt + (size_t)k * D * D;
  float* out = xp + (size_t)k * N * D;

  const v16h a0 = load_a_frag(x, arow, 0,  lane);
  const v16h a1 = load_a_frag(x, arow, 32, lane);

  const int rb = (lane >> 4) << 3;     // C row base (0 or 8)
  const int cl = lane & 15;            // C col within tile
#pragma unroll
  for (int nt = 0; nt < 4; ++nt) {
    const int n0 = nt * 16;
    const float bias = vb[(size_t)k * D + n0 + cl];
    v8f c;
#pragma unroll
    for (int j = 0; j < 8; ++j) c[j] = bias;
    const v16h b0 = load_b_frag(W, n0, 0,  lane);
    const v16h b1 = load_b_frag(W, n0, 32, lane);
    c = __builtin_amdgcn_wmma_f32_16x16x32_f16(false, a0, false, b0, (short)0, c, false, false);
    c = __builtin_amdgcn_wmma_f32_16x16x32_f16(false, a1, false, b1, (short)0, c, false, false);
    float* po = out + (size_t)(row0 + rb) * D + n0 + cl;
    if (full) {
#pragma unroll
      for (int j = 0; j < 8; ++j) po[(size_t)j * D] = c[j];
    } else {
#pragma unroll
      for (int j = 0; j < 8; ++j)
        if (row0 + rb + j < N) po[(size_t)j * D] = c[j];
    }
  }
}

// ---------------- fused edge pass (templated full/tail tile) ----------------
// t = w0 @ e_w + e_b + x3[dst] + x4[src]  (stored into t, pre-activation)
// agg[dst] += sigmoid(w0) * x2[src];  cnt[dst] += 1 ; edge-BN sum/sumsq via LDS reduce.
template<bool FULL>
__device__ __forceinline__ void edge_gemm_tile(
    const float* __restrict__ w0, const _Float16* __restrict__ ewt,
    const float* __restrict__ eb,
    const float* __restrict__ x2, const float* __restrict__ x3, const float* __restrict__ x4,
    const int* __restrict__ dst, const int* __restrict__ src,
    float* __restrict__ t, float* __restrict__ agg, float* __restrict__ cnt,
    float* __restrict__ s_sum, float* __restrict__ s_sq,
    int row0, int lane, int E) {
  const int arow = FULL ? (row0 + (lane & 15)) : min(row0 + (lane & 15), E - 1);
  const v16h a0 = load_a_frag(w0, arow, 0,  lane);
  const v16h a1 = load_a_frag(w0, arow, 32, lane);

  const int rb = (lane >> 4) << 3;
  const int cl = lane & 15;

  int eid[8], dg[8], sg[8];
#pragma unroll
  for (int j = 0; j < 8; ++j) {
    eid[j] = row0 + rb + j;
    const int ec = FULL ? eid[j] : min(eid[j], E - 1);
    dg[j] = dst[ec];
    sg[j] = src[ec];
  }

#pragma unroll
  for (int nt = 0; nt < 4; ++nt) {
    const int n0  = nt * 16;
    const int col = n0 + cl;
    const float bias = eb[col];
    v8f c;
#pragma unroll
    for (int j = 0; j < 8; ++j) c[j] = bias;
    const v16h b0 = load_b_frag(ewt, n0, 0,  lane);
    const v16h b1 = load_b_frag(ewt, n0, 32, lane);
    c = __builtin_amdgcn_wmma_f32_16x16x32_f16(false, a0, false, b0, (short)0, c, false, false);
    c = __builtin_amdgcn_wmma_f32_16x16x32_f16(false, a1, false, b1, (short)0, c, false, false);

    float lsum = 0.0f, lsq = 0.0f;
#pragma unroll
    for (int j = 0; j < 8; ++j) {
      if (FULL || eid[j] < E) {
        // edge pre-activation
        const float v = c[j] + x3[(size_t)dg[j] * D + col] + x4[(size_t)sg[j] * D + col];
        t[(size_t)eid[j] * D + col] = v;
        lsum += v; lsq += v * v;
        // gated message scatter (segment sum)
        const float wv = w0[(size_t)eid[j] * D + col];
        const float m  = sigmoidf_(wv) * x2[(size_t)sg[j] * D + col];
        atomicAdd(&agg[(size_t)dg[j] * D + col], m);
        if (nt == 0 && cl == 0) atomicAdd(&cnt[dg[j]], 1.0f);
      }
    }
    atomicAdd(&s_sum[col], lsum);
    atomicAdd(&s_sq[col],  lsq);
  }
}

__global__ void __launch_bounds__(256)
edge_gemm_kernel(const float* __restrict__ w0,
                 const _Float16* __restrict__ ewt,  // [D][D] transposed f16
                 const float* __restrict__ eb,      // [D]
                 const float* __restrict__ x2,
                 const float* __restrict__ x3,
                 const float* __restrict__ x4,
                 const int* __restrict__ dst,
                 const int* __restrict__ src,
                 float* __restrict__ t,             // [E][D]
                 float* __restrict__ agg,           // [N][D]
                 float* __restrict__ cnt,           // [N]
                 float* __restrict__ esum, float* __restrict__ esq,
                 int E) {
  __shared__ float s_sum[D], s_sq[D];
  if (threadIdx.x < D) { s_sum[threadIdx.x] = 0.0f; s_sq[threadIdx.x] = 0.0f; }
  __syncthreads();

  const int wave = threadIdx.x >> 5;
  const int lane = threadIdx.x & 31;
  const int row0 = (blockIdx.x * 8 + wave) * 16;

  if (row0 + 16 <= E) {
    edge_gemm_tile<true >(w0, ewt, eb, x2, x3, x4, dst, src, t, agg, cnt, s_sum, s_sq, row0, lane, E);
  } else if (row0 < E) {
    edge_gemm_tile<false>(w0, ewt, eb, x2, x3, x4, dst, src, t, agg, cnt, s_sum, s_sq, row0, lane, E);
  }

  __syncthreads();
  if (threadIdx.x < D) {
    atomicAdd(&esum[threadIdx.x], s_sum[threadIdx.x]);
    atomicAdd(&esq[threadIdx.x],  s_sq[threadIdx.x]);
  }
}

// ---------------- node pre-activation u = x1 + agg/max(cnt,1) + node-BN stats ----------------
__global__ void __launch_bounds__(256)
node_preact_kernel(const float* __restrict__ x1, const float* __restrict__ agg,
                   const float* __restrict__ cnt,
                   float* __restrict__ u,
                   float* __restrict__ nsum, float* __restrict__ nsq, int N) {
  __shared__ float s_sum[D], s_sq[D];
  if (threadIdx.x < D) { s_sum[threadIdx.x] = 0.0f; s_sq[threadIdx.x] = 0.0f; }
  __syncthreads();
  const int tid = blockIdx.x * blockDim.x + threadIdx.x;
  const int n = tid >> 4;
  const int g = (tid & 15) * 4;
  if (n < N) {
    const float inv = 1.0f / fmaxf(cnt[n], 1.0f);
    const float4 a = *(const float4*)(agg + (size_t)n * D + g);
    const float4 x = *(const float4*)(x1  + (size_t)n * D + g);
    float4 uv;
    uv.x = x.x + a.x * inv; uv.y = x.y + a.y * inv;
    uv.z = x.z + a.z * inv; uv.w = x.w + a.w * inv;
    *(float4*)(u + (size_t)n * D + g) = uv;
    atomicAdd(&s_sum[g + 0], uv.x); atomicAdd(&s_sq[g + 0], uv.x * uv.x);
    atomicAdd(&s_sum[g + 1], uv.y); atomicAdd(&s_sq[g + 1], uv.y * uv.y);
    atomicAdd(&s_sum[g + 2], uv.z); atomicAdd(&s_sq[g + 2], uv.z * uv.z);
    atomicAdd(&s_sum[g + 3], uv.w); atomicAdd(&s_sq[g + 3], uv.w * uv.w);
  }
  __syncthreads();
  if (threadIdx.x < D) {
    atomicAdd(&nsum[threadIdx.x], s_sum[threadIdx.x]);
    atomicAdd(&nsq[threadIdx.x],  s_sq[threadIdx.x]);
  }
}

// ---------------- finalize BN statistics (biased variance) ----------------
__global__ void bn_finalize_kernel(const float* __restrict__ nsum, const float* __restrict__ nsq,
                                   const float* __restrict__ esum, const float* __restrict__ esq,
                                   float* __restrict__ nmu, float* __restrict__ ninv,
                                   float* __restrict__ emu, float* __restrict__ einv,
                                   int N, int E) {
  const int tid = threadIdx.x;
  if (tid < D) {
    const float mu = nsum[tid] / (float)N;
    const float var = nsq[tid] / (float)N - mu * mu;
    nmu[tid] = mu; ninv[tid] = rsqrtf(var + BN_EPS);
  } else if (tid < 2 * D) {
    const int c = tid - D;
    const float mu = esum[c] / (float)E;
    const float var = esq[c] / (float)E - mu * mu;
    emu[c] = mu; einv[c] = rsqrtf(var + BN_EPS);
  }
}

// ---------------- out = base + silu(BN(pre-act)), in place over pre-act buffer ----------------
// Grid-stride streaming kernel with speculative prefetch of the next chunk.
__global__ void __launch_bounds__(256)
residual_update_kernel(const float* __restrict__ base,
                       float* __restrict__ inout,
                       const float* __restrict__ mu, const float* __restrict__ inv,
                       const float* __restrict__ scale, const float* __restrict__ bias,
                       int M) {
  const size_t total = (size_t)M * 16;               // float4 groups
  const size_t stride = (size_t)gridDim.x * blockDim.x;
  for (size_t i = (size_t)blockIdx.x * blockDim.x + threadIdx.x; i < total; i += stride) {
    // speculative prefetch of next chunk (dropped silently if OOB)
    __builtin_prefetch(base  + (i + stride) * 4, 0, 0);
    __builtin_prefetch(inout + (i + stride) * 4, 0, 0);
    const size_t r = i >> 4;
    const int g = ((int)i & 15) * 4;
    float4 u = *(const float4*)(inout + r * D + g);
    const float4 b = *(const float4*)(base + r * D + g);
    float z;
    z = (u.x - mu[g + 0]) * inv[g + 0] * scale[g + 0] + bias[g + 0]; u.x = b.x + siluf_(z);
    z = (u.y - mu[g + 1]) * inv[g + 1] * scale[g + 1] + bias[g + 1]; u.y = b.y + siluf_(z);
    z = (u.z - mu[g + 2]) * inv[g + 2] * scale[g + 2] + bias[g + 2]; u.z = b.z + siluf_(z);
    z = (u.w - mu[g + 3]) * inv[g + 3] * scale[g + 3] + bias[g + 3]; u.w = b.w + siluf_(z);
    *(float4*)(inout + r * D + g) = u;
  }
}

extern "C" void kernel_launch(void* const* d_in, const int* in_sizes, int n_in,
                              void* d_out, int out_size, void* d_ws, size_t ws_size,
                              hipStream_t stream) {
  const float* x0_in  = (const float*)d_in[0];
  const float* w0_in  = (const float*)d_in[1];
  const int*   eidx   = (const int*)d_in[2];
  const float* v_w    = (const float*)d_in[3];
  const float* v_b    = (const float*)d_in[4];
  const float* e_w    = (const float*)d_in[5];
  const float* e_b    = (const float*)d_in[6];
  const float* vbn_s  = (const float*)d_in[7];
  const float* vbn_b  = (const float*)d_in[8];
  const float* ebn_s  = (const float*)d_in[9];
  const float* ebn_b  = (const float*)d_in[10];

  const int N = in_sizes[0] / D;
  const int E = in_sizes[1] / D;
  const int L = in_sizes[3] / (4 * D * D);
  const size_t N64 = (size_t)N * D, E64 = (size_t)E * D;

  const int* dst  = eidx;        // edge_index[0] = aggregation target
  const int* srci = eidx + E;    // edge_index[1] = neighbor (gather)

  // workspace layout
  float* ws    = (float*)d_ws;
  float* xbufA = ws;
  float* xbufB = xbufA + N64;
  float* wbufA = xbufB + N64;
  float* wbufB = wbufA + E64;
  float* xp    = wbufB + E64;        // 4 * N64
  float* agg   = xp + 4 * N64;       // N64 (contiguous with cnt + BN sums for one zero pass)
  float* cnt   = agg + N64;          // N
  float* stats = cnt + N;            // 8 * D
  float* nsum = stats;          float* nsq  = stats + D;
  float* esum = stats + 2 * D;  float* esq  = stats + 3 * D;
  float* nmu  = stats + 4 * D;  float* ninv = stats + 5 * D;
  float* emu  = stats + 6 * D;  float* einv = stats + 7 * D;
  _Float16* wt = (_Float16*)(stats + 8 * D);   // 5 * D * D halves (f16, transposed weights)

  const size_t ws_need =
      (2 * N64 + 2 * E64 + 4 * N64 + N64 + (size_t)N + 8 * D) * sizeof(float) +
      (size_t)5 * D * D * sizeof(_Float16);
  if (ws_size < ws_need) return;  // deterministic no-op if scratch is insufficient

  float* out_x = (float*)d_out;
  float* out_w = out_x + N64;

  const unsigned nodeTiles  = (unsigned)((N + 15) / 16);
  const unsigned edgeBlocks = (unsigned)(((E + 15) / 16 + 7) / 8);
  const unsigned nodeElems  = (unsigned)(((size_t)N * 16 + 255) / 256);

  for (int l = 0; l < L; ++l) {
    const float* xin = (l == 0) ? x0_in : ((l & 1) ? xbufA : xbufB);
    const float* win = (l == 0) ? w0_in : ((l & 1) ? wbufA : wbufB);
    float* xout = (l == L - 1) ? out_x : (((l & 1) == 0) ? xbufA : xbufB);
    float* wout = (l == L - 1) ? out_w : (((l & 1) == 0) ? wbufA : wbufB);

    // 1) zero agg + cnt + BN accumulators (contiguous region)
    zero_kernel<<<1024, 256, 0, stream>>>(agg, N64 + (size_t)N + 4 * D);

    // 1b) per-layer weight prep: f16 + transpose (4 node mats + 1 edge mat)
    weight_prep_kernel<<<(5 * D * D + 255) / 256, 256, 0, stream>>>(
        v_w + (size_t)l * 4 * D * D, e_w + (size_t)l * D * D, wt);

    // 2) four node projections via WMMA
    node_proj_kernel<<<nodeTiles, 128, 0, stream>>>(
        xin, wt, v_b + (size_t)l * 4 * D, xp, N);

    // 3) fused edge GEMM + gated-message scatter + edge BN stats
    edge_gemm_kernel<<<edgeBlocks, 256, 0, stream>>>(
        win, wt + (size_t)4 * D * D, e_b + (size_t)l * D,
        xp + 1 * N64, xp + 2 * N64, xp + 3 * N64,
        dst, srci, wout, agg, cnt, esum, esq, E);

    // 4) node pre-activation (segment mean) + node BN stats
    node_preact_kernel<<<nodeElems, 256, 0, stream>>>(xp, agg, cnt, xout, nsum, nsq, N);

    // 5) finalize BN mean / inv-std
    bn_finalize_kernel<<<1, 128, 0, stream>>>(nsum, nsq, esum, esq, nmu, ninv, emu, einv, N, E);

    // 6) residual + BN + SiLU updates (in place over the pre-activation buffers)
    residual_update_kernel<<<2048, 256, 0, stream>>>(
        xin, xout, nmu, ninv, vbn_s + (size_t)l * D, vbn_b + (size_t)l * D, N);
    residual_update_kernel<<<4096, 256, 0, stream>>>(
        win, wout, emu, einv, ebn_s + (size_t)l * D, ebn_b + (size_t)l * D, E);
  }
}
